// AttentionReduce_15848429322894
// MI455X (gfx1250) — matrix operand order repaired and checked by reading, not compiled, not run on hardware
//
#include <hip/hip_runtime.h>
#include <hip/hip_bf16.h>

#define NPAT  64
#define MSEQ  16384
#define EDIM  64
#define UNITS 64
#define KSEL  16
#define TPW   4                       // tiles per wave per block

typedef _Float16 v16h __attribute__((ext_vector_type(16)));
typedef _Float16 v8h  __attribute__((ext_vector_type(8)));
typedef float    v8f  __attribute__((ext_vector_type(8)));

// Build a 16-half WMMA operand from two runs of 8 f32 in LDS (b128 loads + cvt).
__device__ __forceinline__ v16h frag_from_f32(const float* p0, const float* p1) {
    float4 q0 = *(const float4*)p0;
    float4 q1 = *(const float4*)(p0 + 4);
    float4 q2 = *(const float4*)p1;
    float4 q3 = *(const float4*)(p1 + 4);
    v16h r;
    r[0]=(_Float16)q0.x;  r[1]=(_Float16)q0.y;  r[2]=(_Float16)q0.z;  r[3]=(_Float16)q0.w;
    r[4]=(_Float16)q1.x;  r[5]=(_Float16)q1.y;  r[6]=(_Float16)q1.z;  r[7]=(_Float16)q1.w;
    r[8]=(_Float16)q2.x;  r[9]=(_Float16)q2.y;  r[10]=(_Float16)q2.z; r[11]=(_Float16)q2.w;
    r[12]=(_Float16)q3.x; r[13]=(_Float16)q3.y; r[14]=(_Float16)q3.z; r[15]=(_Float16)q3.w;
    return r;
}

// Concatenate two 8-half LDS b128 loads into one 16-half WMMA operand.
__device__ __forceinline__ v16h frag_cat(const _Float16* p0, const _Float16* p1) {
    v8h lo = *(const v8h*)p0;
    v8h hi = *(const v8h*)p1;
    v16h r;
#pragma unroll
    for (int i = 0; i < 8; ++i) { r[i] = lo[i]; r[8 + i] = hi[i]; }
    return r;
}

// Branchless SELU: lam*max(x,0) + lam*alpha*(exp(min(x,0))-1)
__device__ __forceinline__ float selu_f(float x) {
    const float lam = 1.0507009873554805f;
    const float la  = 1.7580993408473766f;   // lam * alpha
    return lam * fmaxf(x, 0.0f) + la * (__expf(fminf(x, 0.0f)) - 1.0f);
}

// Compiler-only memory fence: wave-private LDS ops execute in order in HW.
__device__ __forceinline__ void sched_fence() { asm volatile("" ::: "memory"); }

// ---------------------------------------------------------------------------
// Kernel 1: attention scores atts[n,m] = MLP(embed[n,m]) + log(counts[n,m])
// 8 waves/block; each wave owns TPW 16-row tiles; f16 WMMA 16x16x32, f32 acc.
// X tiles streamed HBM->LDS with async-to-LDS copies, double buffered.
// ---------------------------------------------------------------------------
__global__ void __launch_bounds__(256)
scores_kernel(const float* __restrict__ embed, const float* __restrict__ counts,
              const float* __restrict__ W1, const float* __restrict__ b1,
              const float* __restrict__ W2, const float* __restrict__ b2,
              const float* __restrict__ Wo, const float* __restrict__ bo,
              float* __restrict__ atts)
{
    // Weights transposed (sW[n][k]) so B fragments are two runs of 8 contiguous halfs.
    __shared__ __align__(16) _Float16 sW1[UNITS * 72];
    __shared__ __align__(16) _Float16 sW2[UNITS * 72];
    __shared__ float sB1[UNITS], sB2[UNITS], sWo[UNITS];
    __shared__ __align__(16) float    xtf[2 * 8 * 16 * 68]; // double-buffered X (f32)
    __shared__ __align__(16) _Float16 ht[8 * 16 * 72];      // per-wave H tile, f16
    __shared__ float attsrow[8][16];

    const int t    = threadIdx.x;
    const int w    = t >> 5;       // wave in block
    const int l    = t & 31;       // lane
    const int lo16 = l & 15;
    const int hi   = l >> 4;

    // 32 blocks per patient; block covers 512 rows; wave tile stride 128.
    const int n      = blockIdx.x >> 5;
    const int blk_m0 = (blockIdx.x & 31) << 9;

    const float* embase = embed + ((size_t)n * MSEQ + blk_m0 + (w << 4)) * EDIM;

    // Per-wave LDS byte offsets of the two X buffers (low 32 bits of flat addr
    // of an LDS object == its offset in the workgroup LDS segment).
    const uint32_t xb0 = (uint32_t)(uintptr_t)(xtf + w * 1088);
    const uint32_t xb1 = (uint32_t)(uintptr_t)(xtf + 8704 + w * 1088);

    // Issue async HBM->LDS copy of tile `it2` into buffer base `xb`.
    auto issue_tile = [&](int it2, uint32_t xb) {
        const float* xg = embase + (size_t)(it2 << 7) * EDIM;
#pragma unroll
        for (int j = 0; j < 8; ++j) {
            int idx = j * 32 + l;            // float4 slot within 16x64 tile
            int r = idx >> 4;
            int c = (idx & 15) << 2;
            asm volatile("global_load_async_to_lds_b128 %0, %1, off"
                         :: "v"(xb + (uint32_t)((r * 68 + c) * 4)),
                            "v"((unsigned long long)(uintptr_t)(xg + idx * 4))
                         : "memory");
        }
    };

    issue_tile(0, xb0);              // prefetch tile 0 while weights stage

    // Stage W1/W2 transposed as f16 into LDS (4096 elems each, 16 per thread).
#pragma unroll
    for (int i = 0; i < 16; ++i) {
        int idx = i * 256 + t;
        int k = idx >> 6, nn = idx & 63;
        sW1[nn * 72 + k] = (_Float16)W1[idx];
        sW2[nn * 72 + k] = (_Float16)W2[idx];
    }
    if (t < UNITS) { sB1[t] = b1[t]; sB2[t] = b2[t]; sWo[t] = Wo[t]; }
    __syncthreads();

    // Per-lane invariants (bias / output-weight / LDS base pointers).
    float bias1[4], bias2[4], wov[4];
#pragma unroll
    for (int nb = 0; nb < 4; ++nb) {
        bias1[nb] = sB1[nb * 16 + lo16];
        bias2[nb] = sB2[nb * 16 + lo16];
        wov[nb]   = sWo[nb * 16 + lo16];
    }
    const float bo0 = bo[0];
    const float*    xrow0 = xtf + w * 1088 + lo16 * 68 + hi * 8;          // A base (X) buf0
    const float*    xrow1 = xtf + 8704 + w * 1088 + lo16 * 68 + hi * 8;   // A base (X) buf1
    const _Float16* hrow  = ht + w * 1152 + lo16 * 72 + hi * 8;           // A base (H)
    const _Float16* wb1   = sW1 + lo16 * 72 + hi * 8;                     // B base (W1)
    const _Float16* wb2   = sW2 + lo16 * 72 + hi * 8;                     // B base (W2)
    _Float16* hstore = ht + w * 1152 + (8 * hi) * 72 + lo16;              // C store base

    for (int it = 0; it < TPW; ++it) {
        const int m0 = blk_m0 + (it << 7) + (w << 4);

        // Pipeline: issue next tile's async copy, then gate on current tile.
        if (it + 1 < TPW) {
            issue_tile(it + 1, (it & 1) ? xb0 : xb1);
            asm volatile("s_wait_asynccnt 8" ::: "memory");
        } else {
            asm volatile("s_wait_asynccnt 0" ::: "memory");
        }

        // ----- Layer 1: H1 = selu(X @ W1 + b1) -----
        const float* xr = (it & 1) ? xrow1 : xrow0;
        v16h a0 = frag_from_f32(xr,      xr + 16);   // K 0..31
        v16h a1 = frag_from_f32(xr + 32, xr + 48);   // K 32..63
#pragma unroll
        for (int nb = 0; nb < 4; ++nb) {
            v16h bm0 = frag_cat(wb1 + nb * 1152,      wb1 + nb * 1152 + 16);
            v16h bm1 = frag_cat(wb1 + nb * 1152 + 32, wb1 + nb * 1152 + 48);
            v8f c;
#pragma unroll
            for (int i = 0; i < 8; ++i) c[i] = bias1[nb];
            c = __builtin_amdgcn_wmma_f32_16x16x32_f16(false, a0, false, bm0, (short)0, c, false, false);
            c = __builtin_amdgcn_wmma_f32_16x16x32_f16(false, a1, false, bm1, (short)0, c, false, false);
#pragma unroll
            for (int v = 0; v < 8; ++v)
                hstore[v * 72 + nb * 16] = (_Float16)selu_f(c[v]);
        }
        sched_fence();   // wave-private LDS; HW keeps DS ops in order

        // ----- Layer 2 + output dot: atts = selu(H1 @ W2 + b2) @ Wo -----
        v16h h0 = frag_cat(hrow,      hrow + 16);
        v16h h1 = frag_cat(hrow + 32, hrow + 48);
        float acc[8] = {0.f, 0.f, 0.f, 0.f, 0.f, 0.f, 0.f, 0.f};
#pragma unroll
        for (int nb = 0; nb < 4; ++nb) {
            v16h bm0 = frag_cat(wb2 + nb * 1152,      wb2 + nb * 1152 + 16);
            v16h bm1 = frag_cat(wb2 + nb * 1152 + 32, wb2 + nb * 1152 + 48);
            v8f c;
#pragma unroll
            for (int i = 0; i < 8; ++i) c[i] = bias2[nb];
            c = __builtin_amdgcn_wmma_f32_16x16x32_f16(false, h0, false, bm0, (short)0, c, false, false);
            c = __builtin_amdgcn_wmma_f32_16x16x32_f16(false, h1, false, bm1, (short)0, c, false, false);
#pragma unroll
            for (int v = 0; v < 8; ++v) acc[v] += selu_f(c[v]) * wov[nb];
        }
        // Butterfly-reduce each row's partial dot across 16 lanes of a half.
#pragma unroll
        for (int v = 0; v < 8; ++v) {
            float s = acc[v];
            s += __shfl_xor(s, 1, 16);
            s += __shfl_xor(s, 2, 16);
            s += __shfl_xor(s, 4, 16);
            s += __shfl_xor(s, 8, 16);
            if (lo16 == 0) attsrow[w][v + 8 * hi] = s;
        }
        sched_fence();
        if (l < 16) {
            int m = m0 + l;
            float a = attsrow[w][l] + bo0 + __logf(counts[(size_t)n * MSEQ + m]);
            atts[(size_t)n * MSEQ + m] = a;
        }
        sched_fence();
    }
}

// ---------------------------------------------------------------------------
// Kernel 2: per-patient top-16, logsumexp weights, weighted embed reduction.
// ---------------------------------------------------------------------------
__global__ void __launch_bounds__(256)
topk_kernel(const float* __restrict__ atts, const float* __restrict__ counts,
            const float* __restrict__ embed, float* __restrict__ out)
{
    __shared__ float av[64 * 256];       // all 16384 scores of this patient
    __shared__ float sv[256];
    __shared__ int   si[256];
    __shared__ float sred[256];
    __shared__ float topv[KSEL];
    __shared__ int   topi[KSEL];

    const int n = blockIdx.x;
    const int t = threadIdx.x;
    const float NEG = -3.0e38f;

    float bestv = NEG; int besti = 0;
    float csum = 0.f;
#pragma unroll 4
    for (int j = 0; j < 64; ++j) {
        int m = j * 256 + t;
        float a = atts[(size_t)n * MSEQ + m];
        av[j * 256 + t] = a;
        if (a > bestv) { bestv = a; besti = m; }
        csum += counts[(size_t)n * MSEQ + m];
    }
    sred[t] = csum;
    __syncthreads();
    for (int s = 128; s > 0; s >>= 1) {
        if (t < s) sred[t] += sred[t + s];
        __syncthreads();
    }
    float logS = __logf(sred[0]);   // logsumexp(log c) == log(sum c)

    // 16 rounds of incremental block argmax; only the winner's owner rescans.
    for (int it = 0; it < KSEL; ++it) {
        sv[t] = bestv; si[t] = besti;
        __syncthreads();
        for (int s = 128; s > 0; s >>= 1) {
            if (t < s) {
                float ov = sv[t + s]; int oi = si[t + s];
                if (ov > sv[t] || (ov == sv[t] && oi < si[t])) { sv[t] = ov; si[t] = oi; }
            }
            __syncthreads();
        }
        float gv = sv[0]; int gi = si[0];
        if (t == 0) { topv[it] = gv; topi[it] = gi; }
        __syncthreads();
        if ((gi & 255) == t) {                 // owner removes + rescans its 64
            av[(gi >> 8) * 256 + t] = NEG;
            bestv = NEG; besti = 0;
            for (int j = 0; j < 64; ++j) {
                float a = av[j * 256 + t];
                if (a > bestv) { bestv = a; besti = j * 256 + t; }
            }
        }
        __syncthreads();
    }

    // logsumexp over the 16 winners (order-invariant, matches top_k reduction)
    float mx = NEG;
#pragma unroll
    for (int j = 0; j < KSEL; ++j) mx = fmaxf(mx, topv[j]);
    float se = 0.f;
#pragma unroll
    for (int j = 0; j < KSEL; ++j) se += __expf(topv[j] - mx);
    float lse = __logf(se) + mx;

    if (t < EDIM) {
        float acc = 0.f;
#pragma unroll
        for (int j = 0; j < KSEL; ++j) {
            float wgt = __expf(topv[j] - lse);
            acc += wgt * embed[((size_t)n * MSEQ + topi[j]) * EDIM + t];
        }
        out[n * EDIM + t] = acc;
    }
    if (t == 0) out[NPAT * EDIM + n] = lse - logS;   // fold in prior normalizer
}

extern "C" void kernel_launch(void* const* d_in, const int* in_sizes, int n_in,
                              void* d_out, int out_size, void* d_ws, size_t ws_size,
                              hipStream_t stream)
{
    (void)in_sizes; (void)n_in; (void)out_size; (void)ws_size;
    const float* embed  = (const float*)d_in[0];
    const float* counts = (const float*)d_in[1];
    const float* W1     = (const float*)d_in[2];
    const float* b1     = (const float*)d_in[3];
    const float* W2     = (const float*)d_in[4];
    const float* b2     = (const float*)d_in[5];
    const float* Wo     = (const float*)d_in[6];
    const float* bo     = (const float*)d_in[7];
    float* out  = (float*)d_out;
    float* atts = (float*)d_ws;    // N*M floats = 4 MB scratch

    scores_kernel<<<dim3(NPAT * MSEQ / (8 * 16 * TPW)), dim3(256), 0, stream>>>(
        embed, counts, W1, b1, W2, b2, Wo, bo, atts);
    topk_kernel<<<dim3(NPAT), dim3(256), 0, stream>>>(atts, counts, embed, out);
}
